// Sup_ContrastiveCFLoss_19164144075593
// MI455X (gfx1250) — matrix operand order repaired and checked by reading, not compiled, not run on hardware
//
#include <hip/hip_runtime.h>

// MI455X (gfx1250, wave32) fused supervised-contrastive loss.
// Phase 1: L2-normalize fp32 rows -> fp16 features in d_ws.
// Phase 2: register-pipelined f16 WMMA GEMM (sim = F F^T / 0.1) fused with
//          exp/mask epilogue. Column sweep split 4 ways (grid.z) for
//          occupancy; additive per-row partials (Z, S, M) written to unique
//          workspace slots (deterministic, no atomics).
// Phase 3: single-block finalize: per_inst = log(Z) - S/M, scalar mean.

typedef __attribute__((ext_vector_type(16))) _Float16 v16h;
typedef __attribute__((ext_vector_type(8)))  _Float16 v8h;
typedef __attribute__((ext_vector_type(8)))  float    v8f;

// ---------------------------------------------------------------------------
// Phase 1: per-row L2 normalize (one wave32 per row), fp32 -> fp16.
// ---------------------------------------------------------------------------
__global__ __launch_bounds__(256)
void l2norm_f16_kernel(const float* __restrict__ hs,
                       const float* __restrict__ ht0,
                       const float* __restrict__ ht1,
                       _Float16* __restrict__ feat,   // [3][B][D] fp16
                       int B, int D) {
    const int wave = threadIdx.x >> 5;
    const int lane = threadIdx.x & 31;
    const int row  = blockIdx.x * 8 + wave;
    if (row >= B) return;

    const float* src = (blockIdx.y == 0) ? hs : (blockIdx.y == 1 ? ht0 : ht1);
    const float* p = src + (size_t)row * D;
    _Float16*    q = feat + (size_t)blockIdx.y * B * D + (size_t)row * D;

    float v[8];
    float ss = 0.0f;
#pragma unroll
    for (int i = 0; i < 8; ++i) {
        v[i] = p[lane + 32 * i];
        ss += v[i] * v[i];
    }
#pragma unroll
    for (int off = 1; off < 32; off <<= 1)
        ss += __shfl_xor(ss, off, 32);

    const float scale = 1.0f / fmaxf(sqrtf(ss), 1e-12f);
#pragma unroll
    for (int i = 0; i < 8; ++i)
        q[lane + 32 * i] = (_Float16)(v[i] * scale);
}

// ---------------------------------------------------------------------------
// Phase 2 helpers
// ---------------------------------------------------------------------------
__device__ __forceinline__ void loadB(v16h (&b)[8], const _Float16* bp) {
    // B operand: lane = N, K contiguous per half-wave (k = 16*hi + s, as A); 8 chunks fold into
    // immediate offsets off one base -> one s_clause of global_load_b128.
#pragma unroll
    for (int c = 0; c < 8; ++c)
        b[c] = *(const v16h*)(bp + c * 32);
}

__device__ __forceinline__ v8f wmma8(const v16h (&a)[8], const v16h (&b)[8]) {
    v8f c = {};
#pragma unroll
    for (int k = 0; k < 8; ++k)
        c = __builtin_amdgcn_wmma_f32_16x16x32_f16(
            false, a[k], false, b[k], (short)0, c, false, false);
    return c;
}

// ---------------------------------------------------------------------------
// Phase 2: wave = one 16-row tile x one column chunk (quarter of N columns).
// grid = (rowTiles/8, 2 teachers, 4 column chunks). Chunk c: segment c>>1
// (0 = hs rows, 1 = ht rows), half (c&1) of that segment's column tiles.
// C layout: VGPR v, lanes 0-15: M=v, N=lane; lanes 16-31: M=v+8, N=lane-16.
// ---------------------------------------------------------------------------
__global__ __launch_bounds__(256)
void contrast_kernel(const _Float16* __restrict__ feat,  // [3][B][D] fp16
                     const int* __restrict__ labels,     // [B]
                     float* __restrict__ partZ,          // [2][4][N]
                     float* __restrict__ partS,          // [2][4][N]
                     float* __restrict__ partM,          // [2][4][N]
                     int B, int D) {
    __shared__ int sLab[4096];
    for (int i = threadIdx.x; i < B; i += blockDim.x) sLab[i] = labels[i];
    __syncthreads();

    const int lane    = threadIdx.x & 31;
    const int wave    = threadIdx.x >> 5;
    const int teacher = blockIdx.y;            // 0 or 1
    const int chunk   = blockIdx.z;            // 0..3
    const int seg     = chunk >> 1;            // column segment
    const int N       = 2 * B;

    const _Float16* hsn = feat;                                  // rows [0,B)
    const _Float16* htn = feat + (size_t)(1 + teacher) * B * D;  // rows [B,2B)

    const int rowTile = blockIdx.x * (blockDim.x >> 5) + wave;   // 0 .. N/16-1
    const int rb = rowTile * 16;
    if (rb >= N) return;

    const int m  = lane & 15;   // M for A loads / N for B+C
    const int hi = lane >> 4;   // half-wave select

    // ---- A tile, register resident (contiguous k = 16*hi + s per half-wave, as B).
    v16h a[8];
    {
        const int r = rb + m;
        const _Float16* ap = (r < B) ? (hsn + (size_t)r * D)
                                     : (htn + (size_t)(r - B) * D);
#pragma unroll
        for (int c = 0; c < 8; ++c) {
            const int k0 = c * 32 + hi * 16;
            v8h lo = *(const v8h*)(ap + k0);
            v8h up = *(const v8h*)(ap + k0 + 8);
            a[c] = __builtin_shufflevector(lo, up,
                    0, 1, 2, 3, 4, 5, 6, 7, 8, 9, 10, 11, 12, 13, 14, 15);
        }
    }

    // Row labels for the 8 C-VGPR rows this lane sees (from LDS).
    int rl[8];
#pragma unroll
    for (int v = 0; v < 8; ++v) {
        const int r = rb + v + 8 * hi;
        rl[v] = sLab[(r < B) ? r : (r - B)];
    }

    float accZ[8], accS[8], accM[8];
#pragma unroll
    for (int v = 0; v < 8; ++v) { accZ[v] = 0.f; accS[v] = 0.f; accM[v] = 0.f; }

    const int tilesPerSeg = B / 16;                   // 256
    const int tCount      = tilesPerSeg / 2;          // 128 tiles per chunk
    const int t0          = (chunk & 1) * tCount;     // first tile (in-segment)
    const int tEnd        = t0 + tCount;
    const size_t tileStride = (size_t)16 * D;         // halves per column tile

    // Diagonal (i==j) only hits the in-segment tile where cb == rb.
    const int dSeg  = rowTile - seg * tilesPerSeg;
    const int tdiag = (dSeg >= t0 && dSeg < tEnd) ? dSeg : -1;

    // Fused epilogue on a finished 16x16 sim tile (t = in-segment tile idx).
    auto epi = [&](v8f cacc, int t) {
        const int cl = sLab[(t << 4) + m];            // column label (LDS)
        if (t == tdiag) {                             // wave-uniform branch
#pragma unroll
            for (int v = 0; v < 8; ++v) {
                const float sim = cacc[v] * 10.0f;    // 1/TEMPERATURE
                const bool isdiag = ((v + 8 * hi) == m);
                const bool same   = !isdiag && (rl[v] == cl);
                const float e = __expf(sim);
                accZ[v] += isdiag ? 0.0f : e;
                accS[v] += same ? sim  : 0.0f;
                accM[v] += same ? 1.0f : 0.0f;
            }
        } else {
#pragma unroll
            for (int v = 0; v < 8; ++v) {
                const float sim = cacc[v] * 10.0f;
                const bool same = (rl[v] == cl);
                const float e = __expf(sim);
                accZ[v] += e;
                accS[v] += same ? sim  : 0.0f;
                accM[v] += same ? 1.0f : 0.0f;
            }
        }
    };

    {
        const _Float16* segp = (seg == 0) ? hsn : htn;
        const _Float16* colPtr =
            segp + (size_t)m * D + (size_t)hi * 16 + (size_t)t0 * tileStride;

        v16h bA[8], bB[8];
        loadB(bA, colPtr);                      // prologue: first tile

        for (int t = t0; t < tEnd; t += 2) {
            // L2 warm-up hint a few tiles ahead.
            __builtin_prefetch(colPtr + 4 * tileStride, 0, 0);

            // Load t+1 while computing t.
            loadB(bB, colPtr + tileStride);
            epi(wmma8(a, bA), t);

            // Load t+2 while computing t+1 (clamped on the final pair).
            const _Float16* pn = (t + 2 < tEnd) ? (colPtr + 2 * tileStride)
                                                : colPtr;
            loadB(bA, pn);
            epi(wmma8(a, bB), t + 1);

            colPtr += 2 * tileStride;
        }
    }

    // ---- reduce over the 16 lanes of each half-wave (column groups).
#pragma unroll
    for (int v = 0; v < 8; ++v) {
#pragma unroll
        for (int off = 1; off < 16; off <<= 1) {
            accZ[v] += __shfl_xor(accZ[v], off, 32);
            accS[v] += __shfl_xor(accS[v], off, 32);
            accM[v] += __shfl_xor(accM[v], off, 32);
        }
    }

    if (m == 0) {  // lanes 0 and 16: 8 full row results each, unique slots
        const size_t base = (size_t)(teacher * 4 + chunk) * N;
#pragma unroll
        for (int v = 0; v < 8; ++v) {
            const int r = rb + v + 8 * hi;
            partZ[base + r] = accZ[v];
            partS[base + r] = accS[v];
            partM[base + r] = accM[v];
        }
    }
}

// ---------------------------------------------------------------------------
// Phase 3: deterministic single-block finalize.
// per_inst(t,r) = log(sum_c Z) - (sum_c S)/(sum_c M); loss = mean/(2 teachers).
// ---------------------------------------------------------------------------
__global__ __launch_bounds__(256)
void finalize_kernel(const float* __restrict__ partZ,
                     const float* __restrict__ partS,
                     const float* __restrict__ partM,
                     float* __restrict__ d_out, int N) {
    float part = 0.0f;
    for (int i = threadIdx.x; i < 2 * N; i += blockDim.x) {
        const int t = (i < N) ? 0 : 1;
        const int r = (i < N) ? i : (i - N);
        float Z = 0.f, S = 0.f, M = 0.f;
#pragma unroll
        for (int c = 0; c < 4; ++c) {
            const size_t s = (size_t)(t * 4 + c) * N + r;
            Z += partZ[s]; S += partS[s]; M += partM[s];
        }
        part += __logf(Z) - S / M;
    }
#pragma unroll
    for (int off = 1; off < 32; off <<= 1)
        part += __shfl_xor(part, off, 32);

    __shared__ float red[8];
    const int wv = threadIdx.x >> 5, ln = threadIdx.x & 31;
    if (ln == 0) red[wv] = part;
    __syncthreads();
    if (threadIdx.x == 0) {
        float s = 0.f;
#pragma unroll
        for (int i = 0; i < 8; ++i) s += red[i];
        d_out[0] = s * (1.0f / (2.0f * (float)N));
    }
}

// ---------------------------------------------------------------------------
extern "C" void kernel_launch(void* const* d_in, const int* in_sizes, int n_in,
                              void* d_out, int out_size, void* d_ws, size_t ws_size,
                              hipStream_t stream) {
    const float* hs     = (const float*)d_in[0];
    const float* ht0    = (const float*)d_in[1];
    const float* ht1    = (const float*)d_in[2];
    const int*   labels = (const int*)d_in[3];
    // d_in[4] = num_views (always 2) — baked into the math.

    const int B = in_sizes[3];          // 4096
    const int D = in_sizes[0] / B;      // 256
    const int N = 2 * B;

    // Workspace layout: feat (3*B*D fp16 = 6 MB), then 3 partial arrays
    // [2 teachers][4 chunks][N] fp32 (256 KB each). Every partial slot is
    // written exactly once by phase 2 -> no zero-init needed, deterministic.
    _Float16* feat  = (_Float16*)d_ws;
    float*    partZ = (float*)((char*)d_ws + (size_t)3 * B * D * sizeof(_Float16));
    float*    partS = partZ + (size_t)8 * N;
    float*    partM = partS + (size_t)8 * N;
    float*    out   = (float*)d_out;

    // Phase 1: normalize all three matrices (grid.y selects matrix).
    dim3 g1((B + 7) / 8, 3);
    l2norm_f16_kernel<<<g1, 256, 0, stream>>>(hs, ht0, ht1, feat, B, D);

    // Phase 2: 4096 independent waves = 512 row tiles x 2 teachers x 4 chunks.
    dim3 g2((N / 16) / 8, 2, 4);
    contrast_kernel<<<g2, 256, 0, stream>>>(feat, labels, partZ, partS, partM, B, D);

    // Phase 3: deterministic scalar reduction.
    finalize_kernel<<<1, 256, 0, stream>>>(partZ, partS, partM, out, N);
}
